// Net_59304908423598
// MI455X (gfx1250) — compile-verified, matrix operand order checked by reading, not run on hardware
//
#include <hip/hip_runtime.h>
#include <hip/hip_bf16.h>
#include <stdint.h>

#define BB   8
#define PP   2048
#define KNN  20
#define NCLS 23

typedef __attribute__((ext_vector_type(16))) __bf16 v16bf;
typedef __attribute__((ext_vector_type(8)))  __bf16 v8bf;
typedef __attribute__((ext_vector_type(8)))  float  v8f;
typedef __attribute__((ext_vector_type(4)))  float  v4f;
typedef __attribute__((ext_vector_type(2)))  float  v2f;

static __device__ __forceinline__ __bf16 f2bf(float f) { return (__bf16)f; }

static __device__ __forceinline__ v16bf cat8(v8bf lo, v8bf hi)
{
  return __builtin_shufflevector(lo, hi, 0,1,2,3,4,5,6,7,8,9,10,11,12,13,14,15);
}
static __device__ __forceinline__ v16bf mk_frag(v8f lo, v8f hi)
{
  return cat8(__builtin_convertvector(lo, v8bf), __builtin_convertvector(hi, v8bf));
}
// A/B 16-bit fragment chunks for one lane: channels kb+hi*8+[0..7] and kb+16+hi*8+[0..7]
static __device__ __forceinline__ v16bf load_a_frag(const float* row, int kb, int hi8)
{
  v8f lo = *(const v8f*)(row + kb + hi8);
  v8f hi = *(const v8f*)(row + kb + 16 + hi8);
  return mk_frag(lo, hi);
}
static __device__ __forceinline__ v16bf load_b_frag(const __bf16* wrow, int kb, int hi8)
{
  v8bf lo = *(const v8bf*)(wrow + kb + hi8);
  v8bf hi = *(const v8bf*)(wrow + kb + 16 + hi8);
  return cat8(lo, hi);
}
static __device__ __forceinline__ v4f vmax4(v4f a, v4f b)
{
  v4f r;
  r[0] = fmaxf(a[0], b[0]); r[1] = fmaxf(a[1], b[1]);
  r[2] = fmaxf(a[2], b[2]); r[3] = fmaxf(a[3], b[3]);
  return r;
}

// ---------------------------------------------------------------------------
// Point norms |x|^2
// ---------------------------------------------------------------------------
__global__ void norm_kernel(const float* __restrict__ X, float* __restrict__ norms,
                            int D, int n)
{
  int i = blockIdx.x * blockDim.x + threadIdx.x;
  if (i >= n) return;
  const float* r = X + (size_t)i * D;
  float s = 0.f;
  for (int c = 0; c < D; ++c) s += r[c] * r[c];
  norms[i] = s;
}

// ---------------------------------------------------------------------------
// kNN: gram via V_WMMA_F32_16X16X4_F32; query fragments register-cached;
// distance tile through LDS; lanes 0..15 keep insertion-sorted top-20.
// ---------------------------------------------------------------------------
template<int D>
__global__ __launch_bounds__(32) void knn_kernel(const float* __restrict__ X,
                                                 const float* __restrict__ norms,
                                                 int* __restrict__ idx_out)
{
  constexpr int PD = (D + 3) & ~3;
  constexpr int NF = PD / 4;
  const int b    = blockIdx.x;
  const int q0   = blockIdx.y * 16;
  const int lane = threadIdx.x;
  const int lm   = lane & 15;
  const int hi   = lane >> 4;
  const float* Xb = X + (size_t)b * PP * D;
  const float* Nb = norms + (size_t)b * PP;
  const int qrow = q0 + lm;

  __shared__ float nq[16];
  __shared__ float nc[16];
  __shared__ float dt[16 * 16];

  if (hi == 0) nq[lm] = Nb[qrow];

  // A 16x4 f32 frags: lanes 0-15 hold K={k,k+1}, lanes 16-31 hold K={k+2,k+3}
  v2f aq[NF];
  {
    const float* qp = Xb + (size_t)qrow * D;
    #pragma unroll
    for (int f = 0; f < NF; ++f) {
      const int ka = 4 * f + 2 * hi;
      if constexpr (D % 4 == 0) {
        float2 u = *(const float2*)(qp + ka);
        aq[f][0] = u.x; aq[f][1] = u.y;
      } else {
        aq[f][0] = (ka     < D) ? qp[ka]     : 0.f;
        aq[f][1] = (ka + 1 < D) ? qp[ka + 1] : 0.f;
      }
    }
  }

  float td[KNN]; int ti[KNN];
  #pragma unroll
  for (int t = 0; t < KNN; ++t) { td[t] = 3.0e38f; ti[t] = 0; }

  for (int c0 = 0; c0 < PP; c0 += 16) {
    __syncthreads();                       // dt from previous iter consumed
    if (hi == 0) nc[lm] = Nb[c0 + lm];
    __syncthreads();

    v8f acc;
    #pragma unroll
    for (int r = 0; r < 8; ++r) acc[r] = 0.f;

    const float* cp = Xb + (size_t)(c0 + lm) * D;
    #pragma unroll
    for (int f = 0; f < NF; ++f) {
      const int ka = 4 * f + 2 * hi;
      v2f bm;
      if constexpr (D % 4 == 0) {
        float2 u = *(const float2*)(cp + ka);
        bm[0] = u.x; bm[1] = u.y;
      } else {
        bm[0] = (ka     < D) ? cp[ka]     : 0.f;
        bm[1] = (ka + 1 < D) ? cp[ka + 1] : 0.f;
      }
      acc = __builtin_amdgcn_wmma_f32_16x16x4_f32(false, aq[f], false, bm,
                                                  (short)0, acc, false, false);
    }

    // C layout: VGPR r -> row r + 8*hi, col = lane%16
    #pragma unroll
    for (int r = 0; r < 8; ++r) {
      const int m = r + 8 * hi;
      dt[m * 16 + lm] = nq[m] + nc[lm] - 2.0f * acc[r];
    }
    __syncthreads();

    if (hi == 0) {
      for (int n = 0; n < 16; ++n) {
        const float dv = dt[lm * 16 + n];
        if (dv < td[KNN - 1]) {
          int j = KNN - 1;
          while (j > 0 && td[j - 1] > dv) { td[j] = td[j - 1]; ti[j] = ti[j - 1]; --j; }
          td[j] = dv; ti[j] = c0 + n;
        }
      }
    }
  }

  if (hi == 0) {
    for (int t = 0; t < KNN; ++t)
      idx_out[((size_t)b * PP + qrow) * KNN + t] = ti[t];
  }
}

// ---------------------------------------------------------------------------
// EdgeConv stage 1/2: dense GEMMs.  msg(i,j) = xi@Wtop + xj@Wbot - xi@Wbot, so
//   S = X @ Wtop,  T = X @ Wbot   (one A fragment feeds 4 WMMAs: 2 halves x 2 N-tiles)
// ---------------------------------------------------------------------------
template<int DIN, int OUT>
__global__ __launch_bounds__(32) void edge_gemm_kernel(const float* __restrict__ X,
                                                       const __bf16* __restrict__ WtTop,
                                                       const __bf16* __restrict__ WtBot,
                                                       float* __restrict__ S,
                                                       float* __restrict__ T)
{
  constexpr int KPADH = (DIN + 31) & ~31;
  constexpr int KST   = KPADH / 32;
  const int b    = blockIdx.x;
  const int p0   = blockIdx.y * 16;
  const int n0   = blockIdx.z * 32;
  const int lane = threadIdx.x;
  const int lm   = lane & 15;
  const int hi   = lane >> 4;
  const int hi8  = hi * 8;
  const int prow = p0 + lm;
  const float* xi = X + ((size_t)b * PP + prow) * DIN;
  const __bf16* wt0 = WtTop + (size_t)(n0 + lm) * KPADH;
  const __bf16* wt1 = WtTop + (size_t)(n0 + 16 + lm) * KPADH;
  const __bf16* wb0 = WtBot + (size_t)(n0 + lm) * KPADH;
  const __bf16* wb1 = WtBot + (size_t)(n0 + 16 + lm) * KPADH;

  v8f s0, s1, t0, t1;
  #pragma unroll
  for (int r = 0; r < 8; ++r) { s0[r] = 0.f; s1[r] = 0.f; t0[r] = 0.f; t1[r] = 0.f; }

  #pragma unroll
  for (int ks = 0; ks < KST; ++ks) {
    const int kb = ks * 32;
    v16bf a;
    if constexpr (DIN % 32 == 0) {
      a = load_a_frag(xi, kb, hi8);
    } else {
      // DIN == 3: single k-step, guarded scalar loads (weights zero-padded)
      #pragma unroll
      for (int e = 0; e < 16; ++e) {
        const int c = kb + e + ((e & 8) ? 8 : 0) + hi8;
        a[e] = (c < DIN) ? f2bf(xi[c]) : f2bf(0.f);
      }
    }
    const v16bf bt0 = load_b_frag(wt0, kb, hi8);
    const v16bf bt1 = load_b_frag(wt1, kb, hi8);
    const v16bf bb0 = load_b_frag(wb0, kb, hi8);
    const v16bf bb1 = load_b_frag(wb1, kb, hi8);
    s0 = __builtin_amdgcn_wmma_f32_16x16x32_bf16(false, a, false, bt0, (short)0, s0, false, false);
    s1 = __builtin_amdgcn_wmma_f32_16x16x32_bf16(false, a, false, bt1, (short)0, s1, false, false);
    t0 = __builtin_amdgcn_wmma_f32_16x16x32_bf16(false, a, false, bb0, (short)0, t0, false, false);
    t1 = __builtin_amdgcn_wmma_f32_16x16x32_bf16(false, a, false, bb1, (short)0, t1, false, false);
  }

  #pragma unroll
  for (int r = 0; r < 8; ++r) {
    const int m = r + 8 * hi;
    float* sr = S + ((size_t)b * PP + p0 + m) * OUT + lm;
    float* tr = T + ((size_t)b * PP + p0 + m) * OUT + lm;
    sr[n0] = s0[r]; sr[n0 + 16] = s1[r];
    tr[n0] = t0[r]; tr[n0 + 16] = t1[r];
  }
}

// ---------------------------------------------------------------------------
// EdgeConv stage 2/2: gather-max over neighbors (pure bandwidth).
//   Y[p] = S[p] - T[p] + bias + max_k T[idx[p][k]]
// ---------------------------------------------------------------------------
__global__ void emax_kernel(const float* __restrict__ S, const float* __restrict__ T,
                            const int* __restrict__ nidx, const float* __restrict__ bias,
                            float* __restrict__ Y, int OUT)
{
  const int pg  = blockIdx.x;          // 0 .. BB*PP-1
  const int tid = threadIdx.x;         // OUT/4 threads
  const int b   = pg / PP;
  const int* nb = nidx + (size_t)pg * KNN;

  v4f m;
  m[0] = m[1] = m[2] = m[3] = -3.0e38f;
  for (int k = 0; k < KNN; ++k) {
    const int j = nb[k];
    const v4f v = ((const v4f*)(T + ((size_t)b * PP + j) * OUT))[tid];
    m = vmax4(m, v);
  }
  const v4f sv = ((const v4f*)(S + (size_t)pg * OUT))[tid];
  const v4f tv = ((const v4f*)(T + (size_t)pg * OUT))[tid];
  const v4f bv = ((const v4f*)bias)[tid];
  ((v4f*)(Y + (size_t)pg * OUT))[tid] = sv - tv + bv + m;
}

// ---------------------------------------------------------------------------
// fc1 + global max-pool fused: 16x32 bf16 WMMA tile over concat[x1,x2,x3];
// segment-uniform k-steps; column max via __shfl_xor(...,16) (wave32);
// CDNA5 float atomic-max into pooled.
// ---------------------------------------------------------------------------
__global__ __launch_bounds__(32) void fc1_pool_kernel(const float* __restrict__ X1,
                                                      const float* __restrict__ X2,
                                                      const float* __restrict__ X3,
                                                      const __bf16* __restrict__ Wt,
                                                      const float* __restrict__ bias,
                                                      float* __restrict__ pooled)
{
  const int b    = blockIdx.x;
  const int p0   = blockIdx.y * 16;
  const int n0   = blockIdx.z * 32;
  const int lane = threadIdx.x;
  const int lm   = lane & 15;
  const int hi   = lane >> 4;
  const int hi8  = hi * 8;
  const int prow = p0 + lm;
  const float* x1 = X1 + ((size_t)b * PP + prow) * 64;
  const float* x2 = X2 + ((size_t)b * PP + prow) * 128;
  const float* x3 = X3 + ((size_t)b * PP + prow) * 256;
  const __bf16* w0 = Wt + (size_t)(n0 + lm) * 448;
  const __bf16* w1 = Wt + (size_t)(n0 + 16 + lm) * 448;

  v8f acc0, acc1;
  #pragma unroll
  for (int r = 0; r < 8; ++r) { acc0[r] = 0.f; acc1[r] = 0.f; }

  #pragma unroll
  for (int ks = 0; ks < 14; ++ks) {      // 448 = 14*32, segments at 64/192 (mult. of 32)
    const int kb = ks * 32;
    const float* src;
    int cb;
    if (kb < 64)       { src = x1; cb = kb; }
    else if (kb < 192) { src = x2; cb = kb - 64; }
    else               { src = x3; cb = kb - 192; }
    const v16bf a   = load_a_frag(src, cb, hi8);
    const v16bf bf0 = load_b_frag(w0, kb, hi8);
    const v16bf bf1 = load_b_frag(w1, kb, hi8);
    acc0 = __builtin_amdgcn_wmma_f32_16x16x32_bf16(false, a, false, bf0,
                                                   (short)0, acc0, false, false);
    acc1 = __builtin_amdgcn_wmma_f32_16x16x32_bf16(false, a, false, bf1,
                                                   (short)0, acc1, false, false);
  }

  float cm0 = -3.0e38f, cm1 = -3.0e38f;
  #pragma unroll
  for (int r = 0; r < 8; ++r) { cm0 = fmaxf(cm0, acc0[r]); cm1 = fmaxf(cm1, acc1[r]); }
  // column n lives in lane n (rows 0-7) and lane n+16 (rows 8-15)
  cm0 = fmaxf(cm0, __shfl_xor(cm0, 16, 32));
  cm1 = fmaxf(cm1, __shfl_xor(cm1, 16, 32));

  if (hi == 0) {
    float* base = pooled + b * 1024 + n0 + lm;
    const float v0 = cm0 + bias[n0 + lm];
    const float v1 = cm1 + bias[n0 + 16 + lm];
    asm volatile("global_atomic_max_num_f32 %0, %1, off"
                 : : "v"((unsigned long long)(uintptr_t)base), "v"(v0) : "memory");
    asm volatile("global_atomic_max_num_f32 %0, %1, off"
                 : : "v"((unsigned long long)(uintptr_t)(base + 16)), "v"(v1) : "memory");
  }
}

// ---------------------------------------------------------------------------
// Small utility / head kernels
// ---------------------------------------------------------------------------
// Transposed bf16 weights: Wt[n*KPAD + k] = W[(k0+k)*OUT + n] (zero-padded K)
__global__ void wconvT_kernel(const float* __restrict__ W, __bf16* __restrict__ Wt,
                              int k0, int IN, int KPAD, int OUT)
{
  int i = blockIdx.x * blockDim.x + threadIdx.x;
  if (i >= OUT * KPAD) return;
  int n = i / KPAD, k = i % KPAD;
  Wt[i] = (k < IN) ? f2bf(W[(size_t)(k0 + k) * OUT + n]) : f2bf(0.f);
}

__global__ void fill_kernel(float* __restrict__ p, float val, int n)
{
  int i = blockIdx.x * blockDim.x + threadIdx.x;
  if (i < n) p[i] = val;
}

__global__ void head_fc_kernel(const float* __restrict__ in, const float* __restrict__ W,
                               const float* __restrict__ bias, float* __restrict__ out,
                               int IN, int OUT)
{
  int t = blockIdx.x * blockDim.x + threadIdx.x;
  if (t >= BB * OUT) return;
  int b = t / OUT, o = t % OUT;
  float s = bias[o];
  const float* ir = in + (size_t)b * IN;
  for (int i = 0; i < IN; ++i) s += ir[i] * W[(size_t)i * OUT + o];
  out[t] = s;
}

__global__ void bn_relu_kernel(float* __restrict__ x, const float* __restrict__ g,
                               const float* __restrict__ be, int OUT)
{
  int o = blockIdx.x * blockDim.x + threadIdx.x;
  if (o >= OUT) return;
  float mu = 0.f;
  for (int b = 0; b < BB; ++b) mu += x[b * OUT + o];
  mu *= (1.0f / BB);
  float var = 0.f;
  for (int b = 0; b < BB; ++b) { float d = x[b * OUT + o] - mu; var += d * d; }
  var *= (1.0f / BB);
  const float inv = rsqrtf(var + 1e-5f);
  for (int b = 0; b < BB; ++b) {
    float y = g[o] * (x[b * OUT + o] - mu) * inv + be[o];
    x[b * OUT + o] = fmaxf(y, 0.f);
  }
}

__global__ void logsoftmax_kernel(const float* __restrict__ logits, float* __restrict__ out)
{
  int b = threadIdx.x;
  if (b >= BB) return;
  const float* l = logits + b * NCLS;
  float m = -3.0e38f;
  for (int c = 0; c < NCLS; ++c) m = fmaxf(m, l[c]);
  float s = 0.f;
  for (int c = 0; c < NCLS; ++c) s += expf(l[c] - m);
  const float lse = m + logf(s);
  for (int c = 0; c < NCLS; ++c) out[b * NCLS + c] = l[c] - lse;
}

// ---------------------------------------------------------------------------
// Orchestration
// ---------------------------------------------------------------------------
extern "C" void kernel_launch(void* const* d_in, const int* in_sizes, int n_in,
                              void* d_out, int out_size, void* d_ws, size_t ws_size,
                              hipStream_t stream)
{
  (void)in_sizes; (void)n_in; (void)out_size; (void)ws_size;
  const float* pos = (const float*)d_in[0];
  // d_in[1] = batch (repeat(arange(B), P)) -> implicit in layout, unused
  const float* W1  = (const float*)d_in[2];  const float* b1  = (const float*)d_in[3];
  const float* W2  = (const float*)d_in[4];  const float* b2  = (const float*)d_in[5];
  const float* W3  = (const float*)d_in[6];  const float* b3  = (const float*)d_in[7];
  const float* Wf1 = (const float*)d_in[8];  const float* bf1 = (const float*)d_in[9];
  const float* Wa  = (const float*)d_in[10]; const float* ba  = (const float*)d_in[11];
  const float* ga  = (const float*)d_in[12]; const float* bea = (const float*)d_in[13];
  const float* Wb_ = (const float*)d_in[14]; const float* bb_ = (const float*)d_in[15];
  const float* gb  = (const float*)d_in[16]; const float* beb = (const float*)d_in[17];
  const float* Wc  = (const float*)d_in[18]; const float* bc  = (const float*)d_in[19];

  char* w = (char*)d_ws;
  auto carve = [&](size_t bytes) -> void* {
    void* p = (void*)w;
    w += (bytes + 255) & ~size_t(255);
    return p;
  };
  float*  x1     = (float*) carve((size_t)BB * PP * 64  * 4);
  float*  x2     = (float*) carve((size_t)BB * PP * 128 * 4);
  float*  x3     = (float*) carve((size_t)BB * PP * 256 * 4);
  float*  sbuf   = (float*) carve((size_t)BB * PP * 256 * 4);   // S (max OUT=256)
  float*  tbuf   = (float*) carve((size_t)BB * PP * 256 * 4);   // T (max OUT=256)
  int*    idxb   = (int*)   carve((size_t)BB * PP * KNN * 4);
  float*  norms  = (float*) carve((size_t)BB * PP * 4);
  float*  pooled = (float*) carve((size_t)BB * 1024 * 4);
  float*  h1     = (float*) carve((size_t)BB * 512 * 4);
  float*  h2     = (float*) carve((size_t)BB * 256 * 4);
  float*  logits = (float*) carve((size_t)BB * NCLS * 4);
  __bf16* w1tT   = (__bf16*)carve((size_t)64   * 32  * 2);
  __bf16* w1tB   = (__bf16*)carve((size_t)64   * 32  * 2);
  __bf16* w2tT   = (__bf16*)carve((size_t)128  * 64  * 2);
  __bf16* w2tB   = (__bf16*)carve((size_t)128  * 64  * 2);
  __bf16* w3tT   = (__bf16*)carve((size_t)256  * 128 * 2);
  __bf16* w3tB   = (__bf16*)carve((size_t)256  * 128 * 2);
  __bf16* wf1t   = (__bf16*)carve((size_t)1024 * 448 * 2);

  // weight conversion / transpose / padding (fp32 -> bf16, column-major halves)
  auto cv = [&](const float* src, __bf16* dst, int k0, int IN, int KPAD, int OUT) {
    int total = OUT * KPAD;
    wconvT_kernel<<<(total + 255) / 256, 256, 0, stream>>>(src, dst, k0, IN, KPAD, OUT);
  };
  cv(W1,  w1tT, 0,   3,   32,  64);   cv(W1, w1tB, 3,   3,   32,  64);
  cv(W2,  w2tT, 0,   64,  64,  128);  cv(W2, w2tB, 64,  64,  64,  128);
  cv(W3,  w3tT, 0,   128, 128, 256);  cv(W3, w3tB, 128, 128, 128, 256);
  cv(Wf1, wf1t, 0,   448, 448, 1024);

  fill_kernel<<<(BB * 1024 + 255) / 256, 256, 0, stream>>>(pooled, -3.0e38f, BB * 1024);

  const dim3 knng(BB, PP / 16);
  const int  NP = BB * PP;

  // stage 1
  norm_kernel<<<(NP + 255) / 256, 256, 0, stream>>>(pos, norms, 3, NP);
  knn_kernel<3><<<knng, 32, 0, stream>>>(pos, norms, idxb);
  edge_gemm_kernel<3, 64><<<dim3(BB, PP / 16, 64 / 32), 32, 0, stream>>>(pos, w1tT, w1tB, sbuf, tbuf);
  emax_kernel<<<NP, 64 / 4, 0, stream>>>(sbuf, tbuf, idxb, b1, x1, 64);
  // stage 2
  norm_kernel<<<(NP + 255) / 256, 256, 0, stream>>>(x1, norms, 64, NP);
  knn_kernel<64><<<knng, 32, 0, stream>>>(x1, norms, idxb);
  edge_gemm_kernel<64, 128><<<dim3(BB, PP / 16, 128 / 32), 32, 0, stream>>>(x1, w2tT, w2tB, sbuf, tbuf);
  emax_kernel<<<NP, 128 / 4, 0, stream>>>(sbuf, tbuf, idxb, b2, x2, 128);
  // stage 3
  norm_kernel<<<(NP + 255) / 256, 256, 0, stream>>>(x2, norms, 128, NP);
  knn_kernel<128><<<knng, 32, 0, stream>>>(x2, norms, idxb);
  edge_gemm_kernel<128, 256><<<dim3(BB, PP / 16, 256 / 32), 32, 0, stream>>>(x2, w3tT, w3tB, sbuf, tbuf);
  emax_kernel<<<NP, 256 / 4, 0, stream>>>(sbuf, tbuf, idxb, b3, x3, 256);

  // fc1 + global max pool
  fc1_pool_kernel<<<dim3(BB, PP / 16, 1024 / 32), 32, 0, stream>>>(x1, x2, x3, wf1t, bf1, pooled);

  // classifier head
  head_fc_kernel<<<(BB * 512 + 255) / 256, 256, 0, stream>>>(pooled, Wa, ba, h1, 1024, 512);
  bn_relu_kernel<<<(512 + 255) / 256, 256, 0, stream>>>(h1, ga, bea, 512);
  head_fc_kernel<<<(BB * 256 + 255) / 256, 256, 0, stream>>>(h1, Wb_, bb_, h2, 512, 256);
  bn_relu_kernel<<<(256 + 255) / 256, 256, 0, stream>>>(h2, gb, beb, 256);
  head_fc_kernel<<<(BB * NCLS + 255) / 256, 256, 0, stream>>>(h2, Wc, bc, logits, 256, NCLS);
  logsoftmax_kernel<<<1, 32, 0, stream>>>(logits, (float*)d_out);
}
